// FalconAttention_23957327577483
// MI455X (gfx1250) — matrix operand order repaired and checked by reading, not compiled, not run on hardware
//
#include <hip/hip_runtime.h>

// ---------------------------------------------------------------------------
// Falcon attention block for MI455X (gfx1250), bf16 WMMA + fp32 accumulate.
// B=2, S=2048, HID=1024, NH=16, HD=64
// Fragments: 2x b128 per lane. Staging: straight-line b128 loads/stores.
// ---------------------------------------------------------------------------

#define B_   2
#define S_   2048
#define HID_ 1024
#define NH_  16
#define HD_  64
#define ROWS_ (B_ * S_)          // 4096
#define LDP_  40                 // padded LDS row stride (halfs): 80B, 16B-aligned

typedef __attribute__((ext_vector_type(16))) __bf16 v16bf;
typedef __attribute__((ext_vector_type(8)))  float  v8f;

union BF16Frag {
    v16bf  v;
    __bf16 u[16];
    uint4  q[2];
};

union BF16Pair {
    __bf16   h[2];
    unsigned w;
};

static __device__ __forceinline__ v8f wmma_bf16(const BF16Frag& a, const BF16Frag& b, v8f c) {
    return __builtin_amdgcn_wmma_f32_16x16x32_bf16(false, a.v, false, b.v, (short)0, c, false, false);
}

// A-fragment (16x32 bf16) from a row-major tile: lane (g,ln) = two contiguous
// 8-half runs of row ln at k = 8g and 16+8g.
static __device__ __forceinline__ BF16Frag load_afrag_lds(const __bf16* rowp, int g) {
    BF16Frag f;
    f.q[0] = *(const uint4*)(rowp + 8 * g);
    f.q[1] = *(const uint4*)(rowp + 16 + 8 * g);
    return f;
}

// Stage one 16-float run (4x global b128) as 16 bf16 into LDS (2x ds b128).
static __device__ __forceinline__ void stage_row16(const float* src, __bf16* dst) {
    float4 x0 = *(const float4*)(src);
    float4 x1 = *(const float4*)(src + 4);
    float4 x2 = *(const float4*)(src + 8);
    float4 x3 = *(const float4*)(src + 12);
    BF16Frag t;
    t.u[0]  = (__bf16)x0.x; t.u[1]  = (__bf16)x0.y; t.u[2]  = (__bf16)x0.z; t.u[3]  = (__bf16)x0.w;
    t.u[4]  = (__bf16)x1.x; t.u[5]  = (__bf16)x1.y; t.u[6]  = (__bf16)x1.z; t.u[7]  = (__bf16)x1.w;
    t.u[8]  = (__bf16)x2.x; t.u[9]  = (__bf16)x2.y; t.u[10] = (__bf16)x2.z; t.u[11] = (__bf16)x2.w;
    t.u[12] = (__bf16)x3.x; t.u[13] = (__bf16)x3.y; t.u[14] = (__bf16)x3.z; t.u[15] = (__bf16)x3.w;
    *(uint4*)(dst)     = t.q[0];
    *(uint4*)(dst + 8) = t.q[1];
}

// Stage a 2-row x 8-col patch of W into the transposed B tile: k-pairs pack
// into b32 stores.
static __device__ __forceinline__ void stage_bpatch(const float* wp0, const float* wp1,
                                                    __bf16* Bs, int bc, int bk) {
    float4 r0a = *(const float4*)(wp0);
    float4 r0b = *(const float4*)(wp0 + 4);
    float4 r1a = *(const float4*)(wp1);
    float4 r1b = *(const float4*)(wp1 + 4);
    float r0[8] = { r0a.x, r0a.y, r0a.z, r0a.w, r0b.x, r0b.y, r0b.z, r0b.w };
    float r1[8] = { r1a.x, r1a.y, r1a.z, r1a.w, r1b.x, r1b.y, r1b.z, r1b.w };
    #pragma unroll
    for (int c = 0; c < 8; ++c) {
        BF16Pair p;
        p.h[0] = (__bf16)r0[c];
        p.h[1] = (__bf16)r1[c];
        *(unsigned*)&Bs[(bc + c) * LDP_ + bk] = p.w;
    }
}

// ---------------------------------------------------------------------------
// Kernel 1: QKV = X @ Wqkv (bf16 WMMA, fp32 acc) fused with RoPE on Q,K.
// Grid: (3072/64, 4096/64), block 128 (4 waves); wave tile = 16 rows x 64 cols.
// ---------------------------------------------------------------------------
__global__ __launch_bounds__(128) void qkv_rope_kernel(
    const float* __restrict__ X, const float* __restrict__ W,
    __bf16* __restrict__ qw, __bf16* __restrict__ kw, __bf16* __restrict__ vw)
{
    __shared__ __bf16 As[64 * LDP_];   // row-major [row][k], padded
    __shared__ __bf16 Bs[64 * LDP_];   // transposed [col][k], padded

    const int tid  = threadIdx.x;
    const int wave = tid >> 5;
    const int lane = tid & 31;
    const int g    = lane >> 4;
    const int ln   = lane & 15;
    const int col0 = blockIdx.x * 64;
    const int row0 = blockIdx.y * 64;

    // static staging assignments
    const int ar = tid >> 1;            // A: row 0..63
    const int ac = (tid & 1) * 16;      // A: k-offset 0 or 16
    const int bk = (tid >> 3) * 2;      // B: k rows {bk, bk+1}, 0..30
    const int bc = (tid & 7) * 8;       // B: col offset 0..56

    v8f acc[4];
    acc[0] = {}; acc[1] = {}; acc[2] = {}; acc[3] = {};

    for (int k0 = 0; k0 < HID_; k0 += 32) {
        __syncthreads();
        stage_row16(&X[(size_t)(row0 + ar) * HID_ + k0 + ac], &As[ar * LDP_ + ac]);
        stage_bpatch(&W[(size_t)(k0 + bk) * (3 * HID_) + col0 + bc],
                     &W[(size_t)(k0 + bk + 1) * (3 * HID_) + col0 + bc],
                     Bs, bc, bk);
        __builtin_prefetch(&X[(size_t)(row0 + ar) * HID_ + k0 + 32 + ac], 0, 1);
        __builtin_prefetch(&W[(size_t)(k0 + 32 + bk) * (3 * HID_) + col0 + bc], 0, 1);
        __syncthreads();

        const BF16Frag a = load_afrag_lds(&As[(wave * 16 + ln) * LDP_], g);

        #pragma unroll
        for (int nb = 0; nb < 4; ++nb) {
            BF16Frag bf;
            const __bf16* bp = &Bs[(nb * 16 + ln) * LDP_ + 8 * g];
            bf.q[0] = *(const uint4*)(bp);
            bf.q[1] = *(const uint4*)(bp + 16);
            acc[nb] = wmma_bf16(a, bf, acc[nb]);
        }
    }

    // Epilogue: RoPE (Q,K) + store bf16
    const int qkv = col0 >> 10;               // 0=Q 1=K 2=V
    const int h   = (col0 & 1023) >> 6;       // head index

    #pragma unroll
    for (int r = 0; r < 8; ++r) {
        const int m   = r + 8 * g;                 // C-layout row
        const int row = row0 + wave * 16 + m;
        const int s   = row & (S_ - 1);
        const int b   = row >> 11;
        const size_t hb = (size_t)(b * NH_ + h) * S_ * HD_;

        float vals[4];
        if (qkv < 2) {
            #pragma unroll
            for (int half = 0; half < 2; ++half) {
                const int dmod = half * 16 + ln;   // d & 31 (same for d and d+32)
                const float invf = __expf(-0.28782313662425574f * (float)dmod);
                const float ang  = (float)s * invf;
                float sn, c;
                __sincosf(ang, &sn, &c);
                const float xlo = acc[half][r];       // d < 32
                const float xhi = acc[half + 2][r];   // d + 32
                vals[half]     = xlo * c - xhi * sn;
                vals[half + 2] = xhi * c + xlo * sn;
            }
        } else {
            vals[0] = acc[0][r]; vals[1] = acc[1][r];
            vals[2] = acc[2][r]; vals[3] = acc[3][r];
        }

        if (qkv == 0) {
            // Q: row-major [s][d]
            #pragma unroll
            for (int nb = 0; nb < 4; ++nb)
                qw[hb + (size_t)s * HD_ + nb * 16 + ln] = (__bf16)vals[nb];
        } else if (qkv == 1) {
            // K swizzled for B-fragments: ((kt*4 + db)*16 + s%16)*16 + d%16
            #pragma unroll
            for (int nb = 0; nb < 4; ++nb)
                kw[hb + (size_t)((((s >> 4) * 4 + nb) * 16 + (s & 15)) * 16 + ln)] =
                    (__bf16)vals[nb];
        } else {
            // V swizzled for B-fragments: ((vt*64) + d)*16 + s%16
            #pragma unroll
            for (int nb = 0; nb < 4; ++nb)
                vw[hb + (size_t)((((s >> 4) * 64) + nb * 16 + ln) * 16 + (s & 15))] =
                    (__bf16)vals[nb];
        }
    }
}

// ---------------------------------------------------------------------------
// Kernel 2: causal flash attention. One wave per (b,h, 16-query tile).
// Grid: (S/16, B*NH), block 32. All fragments = 2x b128 loads.
// ---------------------------------------------------------------------------
__global__ __launch_bounds__(32) void attn_kernel(
    const __bf16* __restrict__ qw, const __bf16* __restrict__ kw,
    const __bf16* __restrict__ vw, float* __restrict__ ow)
{
    __shared__ __bf16 pbuf[16 * LDP_];

    const int lane = threadIdx.x & 31;
    const int g    = lane >> 4;
    const int ln   = lane & 15;
    const int q0   = blockIdx.x * 16;
    const int bh   = blockIdx.y;          // b*NH + h
    const int b    = bh >> 4;
    const int h    = bh & 15;
    const size_t hb = (size_t)bh * S_ * HD_;

    // Q A-fragments: row-major source, one contiguous 16-half run at k = 16g per fragment
    BF16Frag qa0, qa1;
    {
        const __bf16* qp = qw + hb + (size_t)(q0 + ln) * HD_;
        qa0.q[0] = *(const uint4*)(qp + 16 * g);
        qa0.q[1] = *(const uint4*)(qp + 16 * g + 8);
        qa1.q[0] = *(const uint4*)(qp + 32 + 16 * g);
        qa1.q[1] = *(const uint4*)(qp + 32 + 16 * g + 8);
    }

    float rmax[8], rsum[8];
    #pragma unroll
    for (int r = 0; r < 8; ++r) { rmax[r] = -3.0e38f; rsum[r] = 0.0f; }
    v8f o[4];
    o[0] = {}; o[1] = {}; o[2] = {}; o[3] = {};

    const int jmax = (q0 + 15) >> 5;
    for (int jt = 0; jt <= jmax; ++jt) {
        const int key0 = jt << 5;
        const int kt0  = key0 >> 4;

        // Scores: two 16-key blocks, each = two chained WMMAs over K=64.
        v8f s01[2];
        #pragma unroll
        for (int blk = 0; blk < 2; ++blk) {
            BF16Frag f0, f1;
            const size_t kb  = hb + (size_t)(((kt0 + blk) * 4 + g) * 16 + ln) * 16;
            const size_t kb2 = hb + (size_t)(((kt0 + blk) * 4 + 2 + g) * 16 + ln) * 16;
            f0.q[0] = *(const uint4*)(kw + kb);
            f0.q[1] = *(const uint4*)(kw + kb + 8);
            f1.q[0] = *(const uint4*)(kw + kb2);
            f1.q[1] = *(const uint4*)(kw + kb2 + 8);
            v8f sc = {};
            sc = wmma_bf16(qa0, f0, sc);
            sc = wmma_bf16(qa1, f1, sc);
            s01[blk] = sc;
        }

        const bool need_mask = (key0 + 31) > q0;

        #pragma unroll
        for (int r = 0; r < 8; ++r) {
            const int m = r + 8 * g;
            float a0 = s01[0][r] * 0.125f;     // 1/sqrt(64)
            float a1 = s01[1][r] * 0.125f;
            if (need_mask) {
                if (q0 + m < key0 + ln)      a0 = -3.0e38f;
                if (q0 + m < key0 + 16 + ln) a1 = -3.0e38f;
            }
            float t = fmaxf(a0, a1);
            #pragma unroll
            for (int off = 1; off < 16; off <<= 1)
                t = fmaxf(t, __shfl_xor(t, off, 16));
            const float nm   = fmaxf(rmax[r], t);
            const float corr = __expf(rmax[r] - nm);
            rmax[r] = nm;
            const float p0 = __expf(a0 - nm);
            const float p1 = __expf(a1 - nm);
            float ps = p0 + p1;
            #pragma unroll
            for (int off = 1; off < 16; off <<= 1)
                ps += __shfl_xor(ps, off, 16);
            rsum[r] = rsum[r] * corr + ps;
            o[0][r] *= corr; o[1][r] *= corr; o[2][r] *= corr; o[3][r] *= corr;
            pbuf[m * LDP_ + ln]      = (__bf16)p0;
            pbuf[m * LDP_ + 16 + ln] = (__bf16)p1;
        }
        __syncthreads();

        // P as A-fragment (16x32)
        BF16Frag pa;
        pa.q[0] = *(const uint4*)&pbuf[ln * LDP_ + 16 * g];
        pa.q[1] = *(const uint4*)&pbuf[ln * LDP_ + 16 * g + 8];

        // PV: 4 WMMAs over the 64-wide head dim
        #pragma unroll
        for (int nb = 0; nb < 4; ++nb) {
            BF16Frag vf;
            const size_t vb = hb + (size_t)((kt0 + g) * 64 + nb * 16 + ln) * 16;
            vf.q[0] = *(const uint4*)(vw + vb);
            vf.q[1] = *(const uint4*)(vw + vb + 8);
            o[nb] = wmma_bf16(pa, vf, o[nb]);
        }
        __syncthreads();
    }

    // Finalize: divide by row sums, store fp32 as [B*S, HID]
    #pragma unroll
    for (int r = 0; r < 8; ++r) {
        const float inv = 1.0f / rsum[r];
        const int   m   = r + 8 * g;
        const size_t orow = (size_t)(b * S_ + q0 + m) * HID_ + h * HD_;
        #pragma unroll
        for (int nb = 0; nb < 4; ++nb)
            ow[orow + nb * 16 + ln] = o[nb][r] * inv;
    }
}

// ---------------------------------------------------------------------------
// Kernel 3: out = O @ Wdense + bias. Same tiling as kernel 1.
// ---------------------------------------------------------------------------
__global__ __launch_bounds__(128) void dense_kernel(
    const float* __restrict__ A, const float* __restrict__ W,
    const float* __restrict__ bias, float* __restrict__ out)
{
    __shared__ __bf16 As[64 * LDP_];
    __shared__ __bf16 Bs[64 * LDP_];

    const int tid  = threadIdx.x;
    const int wave = tid >> 5;
    const int lane = tid & 31;
    const int g    = lane >> 4;
    const int ln   = lane & 15;
    const int col0 = blockIdx.x * 64;
    const int row0 = blockIdx.y * 64;

    const int ar = tid >> 1;
    const int ac = (tid & 1) * 16;
    const int bk = (tid >> 3) * 2;
    const int bc = (tid & 7) * 8;

    v8f acc[4];
    acc[0] = {}; acc[1] = {}; acc[2] = {}; acc[3] = {};

    for (int k0 = 0; k0 < HID_; k0 += 32) {
        __syncthreads();
        stage_row16(&A[(size_t)(row0 + ar) * HID_ + k0 + ac], &As[ar * LDP_ + ac]);
        stage_bpatch(&W[(size_t)(k0 + bk) * HID_ + col0 + bc],
                     &W[(size_t)(k0 + bk + 1) * HID_ + col0 + bc],
                     Bs, bc, bk);
        __builtin_prefetch(&A[(size_t)(row0 + ar) * HID_ + k0 + 32 + ac], 0, 1);
        __builtin_prefetch(&W[(size_t)(k0 + 32 + bk) * HID_ + col0 + bc], 0, 1);
        __syncthreads();

        const BF16Frag a = load_afrag_lds(&As[(wave * 16 + ln) * LDP_], g);

        #pragma unroll
        for (int nb = 0; nb < 4; ++nb) {
            BF16Frag bf;
            const __bf16* bp = &Bs[(nb * 16 + ln) * LDP_ + 8 * g];
            bf.q[0] = *(const uint4*)(bp);
            bf.q[1] = *(const uint4*)(bp + 16);
            acc[nb] = wmma_bf16(a, bf, acc[nb]);
        }
    }

    #pragma unroll
    for (int r = 0; r < 8; ++r) {
        const int m   = r + 8 * g;
        const int row = row0 + wave * 16 + m;
        #pragma unroll
        for (int nb = 0; nb < 4; ++nb) {
            const int col = col0 + nb * 16 + ln;
            out[(size_t)row * HID_ + col] = acc[nb][r] + bias[col];
        }
    }
}

// ---------------------------------------------------------------------------
// Host-side launcher
// ---------------------------------------------------------------------------
extern "C" void kernel_launch(void* const* d_in, const int* in_sizes, int n_in,
                              void* d_out, int out_size, void* d_ws, size_t ws_size,
                              hipStream_t stream)
{
    (void)in_sizes; (void)n_in; (void)out_size; (void)ws_size;

    const float* hidden  = (const float*)d_in[0];   // [2,2048,1024]
    const float* w_qkv   = (const float*)d_in[1];   // [1024,3072]
    const float* w_dense = (const float*)d_in[2];   // [1024,1024]
    const float* b_dense = (const float*)d_in[3];   // [1024]
    float*       out     = (float*)d_out;           // [2,2048,1024]

    // Workspace: Q/K/V bf16 (8 MB each) + attn output fp32 (16 MB)
    const size_t QKV_ELEMS = (size_t)B_ * NH_ * S_ * HD_;   // 4,194,304
    __bf16* qws = (__bf16*)d_ws;
    __bf16* kws = qws + QKV_ELEMS;
    __bf16* vws = kws + QKV_ELEMS;
    float*  ows = (float*)(vws + QKV_ELEMS);

    qkv_rope_kernel<<<dim3(3 * HID_ / 64, ROWS_ / 64), 128, 0, stream>>>(
        hidden, w_qkv, qws, kws, vws);

    attn_kernel<<<dim3(S_ / 16, B_ * NH_), 32, 0, stream>>>(
        qws, kws, vws, ows);

    dense_kernel<<<dim3(HID_ / 64, ROWS_ / 64), 128, 0, stream>>>(
        ows, w_dense, b_dense, out);
}
